// Quantize_20091857010713
// MI455X (gfx1250) — matrix operand order repaired and checked by reading, not compiled, not run on hardware
//
#include <hip/hip_runtime.h>
#include <hip/hip_bf16.h>

// ---------------------------------------------------------------------------
// VQ-VAE quantize + EMA codebook update for MI455X (gfx1250, wave32, WMMA).
//   z:            [32,32,32,256] f32  -> flatten [N=32768, D=256]
//   embed:        [D=256, K=8192] f32
//   cluster_size: [K] f32
//   embed_avg:    [D, K] f32
// Outputs (flat): quantize_st [N*D], diff [1], embed_ind [N] (as float),
//                 embed_new [D*K].
// ---------------------------------------------------------------------------

#define N_ROWS  32768
#define DIM     256
#define K_CODES 8192

static constexpr float DECAY  = 0.99f;
static constexpr float EPSV   = 1e-05f;
static constexpr float COMMIT = 0.25f;

typedef __attribute__((ext_vector_type(16))) __bf16 v16bf;
typedef __attribute__((ext_vector_type(8)))  float  v8f;

// ---------------------------------------------------------------------------
// K0: transpose codebook to [K, D] (f32 for exact gather, bf16 for WMMA B
// operand) and compute per-code squared norms. One thread per code.
// ---------------------------------------------------------------------------
__global__ __launch_bounds__(256) void prep_kernel(
    const float* __restrict__ embed,   // [D, K]
    float*       __restrict__ embT32,  // [K, D]
    __bf16*      __restrict__ embT16,  // [K, D]
    float*       __restrict__ cn)      // [K]
{
  const int k = blockIdx.x * 256 + threadIdx.x;
  float acc = 0.0f;
  for (int d = 0; d < DIM; ++d) {
    const float v = embed[(size_t)d * K_CODES + k];   // coalesced across k
    acc += v * v;
    embT32[(size_t)k * DIM + d] = v;
    embT16[(size_t)k * DIM + d] = (__bf16)v;
  }
  cn[k] = acc;
}

__global__ __launch_bounds__(256) void zero_kernel(float* __restrict__ p, int n)
{
  const int i = blockIdx.x * 256 + threadIdx.x;
  if (i < n) p[i] = 0.0f;
}

// ---------------------------------------------------------------------------
// K1: fused distance + argmin via v_wmma_f32_16x16x32_bf16, with the B
// (codebook) stream staged through LDS by the CDNA5 async global->LDS path.
//
// Block = 128 threads = 4 waves; wave w owns rows [blk*128 + w*32, +32) as
// TWO 16-row M-tiles whose A fragments (bf16, ISA 16-bit A 16x32 layout) are
// held in registers for the entire K sweep. K is swept in 32-column
// superblocks; each superblock of B (32 cols x 256 d bf16 = 16 KB) is copied
// global->LDS with global_load_async_to_lds_b128 into a double buffer, and is
// stored ALREADY IN FRAGMENT ORDER: frag[sb][s][lane] is the 32-byte per-lane
// slice of the dense 32x16 B operand (lane<16: col=lane, d=32s..32s+7, 32s+16..32s+23;
// lane>=16: col=lane-16, d=32s+8..32s+15, 32s+24..32s+31). Consumption is then a pair of
// conflict-free ds_load_b128 at (lane*32), identical for all waves, and each
// fragment feeds 2 WMMAs (tile0, tile1).
//
// Pipeline per superblock: s_wait_asynccnt 0 ; barrier ; issue async fill of
// next superblock into the other buffer ; consume current buffer. Double
// buffering makes this hazard-free, and global latency is hidden a full
// iteration (32 WMMAs) ahead.
// ---------------------------------------------------------------------------
__global__ __launch_bounds__(128) void dist_argmin_kernel(
    const float*  __restrict__ z,        // [N, D]
    const __bf16* __restrict__ embT16,   // [K, D]
    const float*  __restrict__ cn,       // [K]
    int*          __restrict__ ind_int,  // [N]
    float*        __restrict__ ind_f)    // [N] (output as float)
{
  __shared__ __align__(32) unsigned char smem[2 * 16384];  // double buffer

  const int lane  = threadIdx.x & 31;
  const int wave  = threadIdx.x >> 5;
  const int rbase = blockIdx.x * 128 + wave * 32;
  const int lc    = lane & 15;
  const bool hihalf = (lane >= 16);

  // ---- load A fragments for both M-tiles into registers (bf16) ----
  const float* rowp0 = z + (size_t)(rbase + lc) * DIM;
  const float* rowp1 = z + (size_t)(rbase + 16 + lc) * DIM;

  v16bf a0[8], a1[8];
#pragma unroll
  for (int s = 0; s < 8; ++s) {
    const int d0 = 32 * s + (hihalf ? 8 : 0);
#pragma unroll
    for (int t = 0; t < 2; ++t) {
      const float* rp = t ? rowp1 : rowp0;
      const float4 f0 = *(const float4*)(rp + d0);
      const float4 f1 = *(const float4*)(rp + d0 + 4);
      const float4 f2 = *(const float4*)(rp + d0 + 16);
      const float4 f3 = *(const float4*)(rp + d0 + 20);
      v16bf av;
      av[0]  = (__bf16)f0.x; av[1]  = (__bf16)f0.y;
      av[2]  = (__bf16)f0.z; av[3]  = (__bf16)f0.w;
      av[4]  = (__bf16)f1.x; av[5]  = (__bf16)f1.y;
      av[6]  = (__bf16)f1.z; av[7]  = (__bf16)f1.w;
      av[8]  = (__bf16)f2.x; av[9]  = (__bf16)f2.y;
      av[10] = (__bf16)f2.z; av[11] = (__bf16)f2.w;
      av[12] = (__bf16)f3.x; av[13] = (__bf16)f3.y;
      av[14] = (__bf16)f3.z; av[15] = (__bf16)f3.w;
      if (t) a1[s] = av; else a0[s] = av;
    }
  }

  // ---- per-thread async-fill unit assignments ----
  // A 32-col superblock = 1024 units of 16 bytes. Unit u: col = u>>5 (0..31),
  // m = u&31 (16B chunk along d). LDS destination is the fragment-order slot:
  //   sub-block sb = col>>4, s = m>>2, r = m&3,
  //   lane' = (col&15) + 16*(r&1), byte off16 = (r>>1)*16.
  // Global source (within embT16): col*512 + m*16 (+ iter*16384).
  // 128 threads x 8 units each; consecutive threads touch consecutive 16B
  // chunks of the same codebook row -> coalesced.
  const unsigned ldsbase = (unsigned)(uintptr_t)(&smem[0]);
  unsigned u_lds[8], u_g[8];
#pragma unroll
  for (int i = 0; i < 8; ++i) {
    const unsigned u   = (unsigned)threadIdx.x + 128u * i;
    const unsigned col = u >> 5;
    const unsigned m   = u & 31u;
    const unsigned sb  = col >> 4;
    const unsigned s   = m >> 2;
    const unsigned r   = m & 3u;
    const unsigned ln  = (col & 15u) + ((r & 1u) ? 16u : 0u);
    const unsigned o16 = (r >> 1u) * 16u;
    u_lds[i] = ldsbase + sb * 8192u + s * 1024u + ln * 32u + o16;
    u_g[i]   = col * 512u + m * 16u;
  }

#define ASYNC_FILL(ITER)                                                      \
  do {                                                                        \
    const unsigned _go = (unsigned)(ITER) * 16384u;                           \
    const unsigned _lo = ((unsigned)(ITER) & 1u) << 14;                       \
    _Pragma("unroll")                                                         \
    for (int _i = 0; _i < 8; ++_i) {                                          \
      const unsigned _l = u_lds[_i] + _lo;                                    \
      const unsigned _g = u_g[_i] + _go;                                      \
      asm volatile("global_load_async_to_lds_b128 %0, %1, %2"                 \
                   :: "v"(_l), "v"(_g), "s"(embT16) : "memory");              \
    }                                                                         \
  } while (0)

  float bestv0[8], bestv1[8];
  int   besti0[8], besti1[8];
#pragma unroll
  for (int j = 0; j < 8; ++j) {
    bestv0[j] = 3.4e38f; besti0[j] = 0;
    bestv1[j] = 3.4e38f; besti1[j] = 0;
  }

  ASYNC_FILL(0);

  for (int it = 0; it < K_CODES / 32; ++it) {
    asm volatile("s_wait_asynccnt 0x0" ::: "memory");
    __syncthreads();                    // all fills for `it` are in LDS
    if (it < K_CODES / 32 - 1) ASYNC_FILL(it + 1);

    const unsigned bufo = ((unsigned)it & 1u) << 14;
#pragma unroll
    for (int sb = 0; sb < 2; ++sb) {
      const int colbase = it * 32 + sb * 16;
      const float cnv = cn[colbase + lc];

      v8f acc0 = {0.f, 0.f, 0.f, 0.f, 0.f, 0.f, 0.f, 0.f};
      v8f acc1 = {0.f, 0.f, 0.f, 0.f, 0.f, 0.f, 0.f, 0.f};
#pragma unroll
      for (int s = 0; s < 8; ++s) {
        const v16bf b = *(const v16bf*)(smem + bufo + sb * 8192 + s * 1024 + lane * 32);
        acc0 = __builtin_amdgcn_wmma_f32_16x16x32_bf16(
            false, a0[s], false, b, (short)0, acc0, false, false);
        acc1 = __builtin_amdgcn_wmma_f32_16x16x32_bf16(
            false, a1[s], false, b, (short)0, acc1, false, false);
      }
#pragma unroll
      for (int j = 0; j < 8; ++j) {
        const float v0 = cnv - 2.0f * acc0[j];
        const float v1 = cnv - 2.0f * acc1[j];
        const int col = colbase + lc;
        if (v0 < bestv0[j]) { bestv0[j] = v0; besti0[j] = col; }
        if (v1 < bestv1[j]) { bestv1[j] = v1; besti1[j] = col; }
      }
    }
  }
#undef ASYNC_FILL

  // Cross-lane argmin reduction within each 16-lane half; tie -> smaller idx.
#pragma unroll
  for (int t = 0; t < 2; ++t) {
#pragma unroll
    for (int j = 0; j < 8; ++j) {
      float v = t ? bestv1[j] : bestv0[j];
      int idx = t ? besti1[j] : besti0[j];
#pragma unroll
      for (int m = 1; m < 16; m <<= 1) {
        const float ov = __shfl_xor(v, m, 32);
        const int   oi = __shfl_xor(idx, m, 32);
        if (ov < v || (ov == v && oi < idx)) { v = ov; idx = oi; }
      }
      if (lc == 0) {
        const int row = rbase + t * 16 + j + (hihalf ? 8 : 0);
        ind_int[row] = idx;
        ind_f[row]   = (float)idx;
      }
    }
  }
}

// ---------------------------------------------------------------------------
// K2: per-row gather of the selected code (exact f32 from transposed copy),
// quantize_st output, per-row diff partial (deterministic), and segment-sum
// scatter (counts are sums of 1.0f -> exactly associative; embed_sum uses
// global_atomic_add_f32).
// ---------------------------------------------------------------------------
__global__ __launch_bounds__(256) void gather_scatter_kernel(
    const float* __restrict__ z,          // [N, D]
    const float* __restrict__ embT32,     // [K, D]
    const int*   __restrict__ ind_int,    // [N]
    float*       __restrict__ q_out,      // [N, D]
    float*       __restrict__ partial,    // [N]
    float*       __restrict__ counts,     // [K]
    float*       __restrict__ embed_sum)  // [D, K]
{
  const int row = blockIdx.x;
  const int d   = threadIdx.x;
  const int k   = ind_int[row];

  const float zv = z[(size_t)row * DIM + d];
  const float qv = embT32[(size_t)k * DIM + d];
  q_out[(size_t)row * DIM + d] = qv;

  float e = qv - zv;
  e *= e;
#pragma unroll
  for (int m = 16; m >= 1; m >>= 1) e += __shfl_xor(e, m, 32);

  __shared__ float sred[8];
  if ((threadIdx.x & 31) == 0) sred[threadIdx.x >> 5] = e;
  __syncthreads();
  if (threadIdx.x == 0) {
    float s = 0.0f;
#pragma unroll
    for (int i = 0; i < 8; ++i) s += sred[i];
    partial[row] = s;
    atomicAdd(&counts[k], 1.0f);
  }
  atomicAdd(&embed_sum[(size_t)d * K_CODES + k], zv);
}

// ---------------------------------------------------------------------------
// K3: deterministic reductions — diff scalar, cluster_size EMA + n = sum.
// ---------------------------------------------------------------------------
__global__ __launch_bounds__(256) void finalize1_kernel(
    const float* __restrict__ partial,       // [N]
    const float* __restrict__ counts,        // [K]
    const float* __restrict__ cluster_size,  // [K]
    float*       __restrict__ csn,           // [K] cluster_size_new
    float*       __restrict__ n_out,         // [1]
    float*       __restrict__ diff_out)      // [1]
{
  const int t = threadIdx.x;
  float ds = 0.0f;
  for (int i = t; i < N_ROWS; i += 256) ds += partial[i];
  float na = 0.0f;
  for (int k = t; k < K_CODES; k += 256) {
    const float c = cluster_size[k] * DECAY + (1.0f - DECAY) * counts[k];
    csn[k] = c;
    na += c;
  }
  __shared__ float s1[256], s2[256];
  s1[t] = ds; s2[t] = na;
  __syncthreads();
  for (int off = 128; off > 0; off >>= 1) {
    if (t < off) { s1[t] += s1[t + off]; s2[t] += s2[t + off]; }
    __syncthreads();
  }
  if (t == 0) {
    diff_out[0] = COMMIT * s1[0] / (float)((size_t)N_ROWS * DIM);
    n_out[0] = s2[0];
  }
}

// ---------------------------------------------------------------------------
// K4: embed_new = (embed_avg*DECAY + (1-DECAY)*embed_sum) / cs_adj(k)
// ---------------------------------------------------------------------------
__global__ __launch_bounds__(256) void finalize2_kernel(
    const float* __restrict__ embed_avg,  // [D, K]
    const float* __restrict__ embed_sum,  // [D, K]
    const float* __restrict__ csn,        // [K]
    const float* __restrict__ n_ptr,      // [1]
    float*       __restrict__ enew)       // [D, K]
{
  const int idx = blockIdx.x * 256 + threadIdx.x;   // < D*K
  const int k = idx & (K_CODES - 1);
  const float n = n_ptr[0];
  const float cs = (csn[k] + EPSV) / (n + (float)K_CODES * EPSV) * n;
  const float avg = embed_avg[idx] * DECAY + (1.0f - DECAY) * embed_sum[idx];
  enew[idx] = avg / cs;
}

// ---------------------------------------------------------------------------
// Host launcher
// ---------------------------------------------------------------------------
extern "C" void kernel_launch(void* const* d_in, const int* in_sizes, int n_in,
                              void* d_out, int out_size, void* d_ws, size_t ws_size,
                              hipStream_t stream) {
  (void)in_sizes; (void)n_in; (void)out_size; (void)ws_size;

  const float* z            = (const float*)d_in[0];
  const float* embed        = (const float*)d_in[1];
  const float* cluster_size = (const float*)d_in[2];
  const float* embed_avg    = (const float*)d_in[3];

  float* out      = (float*)d_out;
  float* q_out    = out;                                   // [N*D]
  float* diff_out = out + (size_t)N_ROWS * DIM;            // [1]
  float* indf_out = diff_out + 1;                          // [N]
  float* enew_out = indf_out + N_ROWS;                     // [D*K]

  // Workspace layout (bytes), ~21.4 MB total.
  char* ws = (char*)d_ws;
  float*  embT32    = (float*) (ws + 0);          //  8,388,608 B  [K,D] f32
  __bf16* embT16    = (__bf16*)(ws + 8388608);    //  4,194,304 B  [K,D] bf16
  float*  cn        = (float*) (ws + 12582912);   //     32,768 B  [K]
  float*  counts    = (float*) (ws + 12615680);   //     32,768 B  [K]
  float*  csn       = (float*) (ws + 12648448);   //     32,768 B  [K]
  float*  embed_sum = (float*) (ws + 12681216);   //  8,388,608 B  [D,K]
  float*  partial   = (float*) (ws + 21069824);   //    131,072 B  [N]
  int*    ind_int   = (int*)   (ws + 21200896);   //    131,072 B  [N]
  float*  n_ws      = (float*) (ws + 21331968);   //          4 B

  // K0: codebook transpose (+bf16) + column norms.
  prep_kernel<<<K_CODES / 256, 256, 0, stream>>>(embed, embT32, embT16, cn);

  // Zero the accumulation buffers (ws is not re-poisoned between replays).
  zero_kernel<<<(K_CODES + 255) / 256, 256, 0, stream>>>(counts, K_CODES);
  zero_kernel<<<((DIM * K_CODES) + 255) / 256, 256, 0, stream>>>(embed_sum, DIM * K_CODES);

  // K1: WMMA distance matmul + fused argmin (async-LDS staged B stream).
  dist_argmin_kernel<<<N_ROWS / 128, 128, 0, stream>>>(z, embT16, cn, ind_int, indf_out);

  // K2: gather quantize, diff partials, segment-sum scatter.
  gather_scatter_kernel<<<N_ROWS, 256, 0, stream>>>(z, embT32, ind_int, q_out,
                                                    partial, counts, embed_sum);

  // K3/K4: reductions + EMA normalize.
  finalize1_kernel<<<1, 256, 0, stream>>>(partial, counts, cluster_size, csn, n_ws, diff_out);
  finalize2_kernel<<<(DIM * K_CODES) / 256, 256, 0, stream>>>(embed_avg, embed_sum, csn, n_ws, enew_out);
}
